// NeighborhoodCrossAttention2D_40054865002933
// MI455X (gfx1250) — matrix operand order repaired and checked by reading, not compiled, not run on hardware
//
#include <hip/hip_runtime.h>

// ---------------------------------------------------------------------------
// Neighborhood cross-attention 2D for MI455X (gfx1250), wave32 + WMMA bf16.
// B=2, H=W=56, C=256, heads=8, hd=32, K=7. Three kernels:
//   1) qkv GEMM  (x @ W^T + b)        -> q(scaled)/k/v bf16 [b,h,pix,32]
//   2) slab attention: per 8x8 pixel tile, K slab staged by the Tensor Data
//      Mover (tensor_load_to_lds, TENSORcnt) with async/scalar fallbacks,
//      QK^T over padded 224-neighbor slab via v_wmma_f32_16x16x32_bf16,
//      masked+biased softmax, P@V via WMMA
//   3) proj GEMM (bf16 A, f32 W)      -> f32 output
// ---------------------------------------------------------------------------

typedef __bf16 bf16;
typedef __attribute__((ext_vector_type(16))) __bf16 v16bf;
typedef __attribute__((ext_vector_type(8)))  __bf16 v8bf;
typedef __attribute__((ext_vector_type(8)))  float  v8f;
typedef __attribute__((ext_vector_type(4)))  float  v4f;

#if defined(__gfx1250__) && __has_builtin(__builtin_amdgcn_tensor_load_to_lds) && \
    __has_builtin(__builtin_amdgcn_s_wait_tensorcnt)
#define HAVE_TDM 1
#else
#define HAVE_TDM 0
#endif

#if defined(__gfx1250__) && __has_builtin(__builtin_amdgcn_global_load_async_to_lds_b128)
#define HAVE_ASYNC_LDS 1
#else
#define HAVE_ASYNC_LDS 0
#endif

#define WMMA_BF16(A, B, C) \
  __builtin_amdgcn_wmma_f32_16x16x32_bf16(false, (A), false, (B), (short)0, (C), false, false)

__device__ __forceinline__ int imin(int a, int b) { return a < b ? a : b; }
__device__ __forceinline__ int imax(int a, int b) { return a > b ? a : b; }

// B-fragment: lane holds n = lane&15, elements e=0..15 map to k = k0 + (lane>>4)*16 + e
// (caller pre-adds k0 + half*16): 16 contiguous bf16.
__device__ __forceinline__ v16bf load16bf(const bf16* p) {
  v8bf lo = *(const v8bf*)p;
  v8bf hi = *(const v8bf*)(p + 8);
  v16bf r;
#pragma unroll
  for (int i = 0; i < 8; i++) { r[i] = lo[i]; r[8 + i] = hi[i]; }
  return r;
}

// A-fragment (16x32 bf16): lane holds m = lane&15; elements e=0..15 -> k = k0+half*16+e.
// Caller pre-adds k0 + half*16.
__device__ __forceinline__ v16bf load_a_bf(const bf16* p) {
  v8bf lo = *(const v8bf*)p;
  v8bf hi = *(const v8bf*)(p + 8);
  v16bf r;
#pragma unroll
  for (int i = 0; i < 8; i++) { r[i] = lo[i]; r[8 + i] = hi[i]; }
  return r;
}

// Same A-fragment but from fp32 source, converting to bf16.
__device__ __forceinline__ v16bf load_a_f32(const float* p) {
  v4f a0 = *(const v4f*)p,        a1 = *(const v4f*)(p + 4);
  v4f a2 = *(const v4f*)(p + 8), a3 = *(const v4f*)(p + 12);
  v16bf r;
#pragma unroll
  for (int i = 0; i < 4; i++) {
    r[i]      = (bf16)a0[i];
    r[4 + i]  = (bf16)a1[i];
    r[8 + i]  = (bf16)a2[i];
    r[12 + i] = (bf16)a3[i];
  }
  return r;
}

// B-fragment from fp32 source: 16 contiguous floats, converted.
__device__ __forceinline__ v16bf load_b_f32(const float* p) {
  v4f a0 = *(const v4f*)p,       a1 = *(const v4f*)(p + 4);
  v4f a2 = *(const v4f*)(p + 8), a3 = *(const v4f*)(p + 12);
  v16bf r;
#pragma unroll
  for (int i = 0; i < 4; i++) {
    r[i]      = (bf16)a0[i];
    r[4 + i]  = (bf16)a1[i];
    r[8 + i]  = (bf16)a2[i];
    r[12 + i] = (bf16)a3[i];
  }
  return r;
}

// ---------------------------------------------------------------------------
// Kernel 1: y = x @ W^T + bias, output layout [b, head, pix, d] bf16.
// grid (392, 16, 3): z=0 -> q (scaled), z=1 -> k, z=2 -> v. One wave / 16x16 tile.
// ---------------------------------------------------------------------------
__global__ __launch_bounds__(32)
void qkv_gemm_kernel(const float* __restrict__ xq, const float* __restrict__ xkv,
                     const float* __restrict__ W, const float* __restrict__ bias,
                     bf16* __restrict__ qb, bf16* __restrict__ kb, bf16* __restrict__ vb) {
  const int lane = threadIdx.x & 31;
  const int nl = lane & 15, half = lane >> 4;
  const int row0 = blockIdx.x * 16;
  const int col0 = blockIdx.y * 16;
  const int which = blockIdx.z;

  const float* A = (which == 0) ? xq : xkv;
  const int wrow = ((which == 2) ? 256 : 0) + col0 + nl;
  const float* brow = W + (size_t)wrow * 256;
  const float* arow = A + (size_t)(row0 + nl) * 256;

  v8f acc = {};
#pragma unroll
  for (int k0 = 0; k0 < 256; k0 += 32) {
    v16bf a = load_a_f32(arow + k0 + half * 16);
    v16bf b = load_b_f32(brow + k0 + half * 16);
    acc = WMMA_BF16(a, b, acc);
  }

  const float bval = bias[wrow];
  const float scale = (which == 0) ? 0.17677669529663687f : 1.0f;  // hd^-0.5
  bf16* outp = (which == 0) ? qb : ((which == 1) ? kb : vb);
  const int h = col0 >> 5;
  const int d = (col0 & 31) + nl;
#pragma unroll
  for (int r = 0; r < 8; r++) {
    int gm = row0 + r + half * 8;          // global row (batch*3136 + pix)
    int bi = gm / 3136;
    int pix = gm - bi * 3136;
    size_t o = ((size_t)(bi * 8 + h) * 3136 + pix) * 32 + d;
    outp[o] = (bf16)((acc[r] + bval) * scale);
  }
}

// ---------------------------------------------------------------------------
// Kernel 2: slab neighborhood attention. 784 blocks (b*8*49), 128 threads.
// ---------------------------------------------------------------------------
#define NN 224  // padded slab size: 14*14=196 -> 224 = 7*32

__global__ __launch_bounds__(128)
void natten_kernel(const bf16* __restrict__ qbuf, const bf16* __restrict__ kbuf,
                   const bf16* __restrict__ vbuf, const float* __restrict__ rpb,
                   bf16* __restrict__ attn_out) {
  extern __shared__ char smem[];
  bf16*  Ks  = (bf16*)smem;               // [NN][32]  K slab, row-major
  bf16*  Vst = Ks + NN * 32;              // [32][NN]  V slab, transposed (d-major)
  float* Ss  = (float*)(Vst + 32 * NN);   // [64][NN]  raw scores
  bf16*  Ps  = (bf16*)(Ss + 64 * NN);     // [64][NN]  probabilities

  const int blk = blockIdx.x;
  const int tile = blk % 49;
  const int h = (blk / 49) & 7;
  const int b = blk / (49 * 8);
  const int x0 = (tile / 7) * 8, y0 = (tile % 7) * 8;
  const int row_lo = imin(imax(x0 - 3, 0), 49);
  const int col_lo = imin(imax(y0 - 3, 0), 49);
  const int tid = threadIdx.x;
  const int wave = tid >> 5, lane = tid & 31;
  const int nl = lane & 15, half = lane >> 4;

  const size_t headoff = (size_t)(b * 8 + h) * 3136 * 32;
  const bf16* kbase = kbuf + headoff;
  const bf16* vbase = vbuf + headoff;
  const bf16* qbase = qbuf + headoff;

  // ---- K slab staging ----------------------------------------------------
#if HAVE_TDM
  // One TDM descriptor per block: 3D tile (z=14 rows, y=14 cols, x=32 ch, 2B)
  // from tensor [56,56,32]. tensor_dim1/2 are set to the distance from the
  // tile origin to the tensor edge so hardware OOB handling zero-fills the
  // clamped-window padding for us.
  if (wave == 0) {
    typedef __attribute__((ext_vector_type(4))) unsigned int v4u_;
    typedef __attribute__((ext_vector_type(8))) int v8i_;
    typedef __attribute__((ext_vector_type(4))) int v4i_;
    unsigned long long ga =
        (unsigned long long)(const void*)(kbase + (size_t)(row_lo * 56 + col_lo) * 32);
    unsigned int lds_off =
        (unsigned int)(unsigned long long)(__attribute__((address_space(3))) char*)(void*)Ks;
    const unsigned int tdim0 = 32u;                       // channel extent
    const unsigned int tdim1 = (unsigned int)(56 - col_lo);
    const unsigned int tdim2 = (unsigned int)(56 - row_lo);
    const unsigned long long s0 = 32ull;                  // stride y  (elements)
    const unsigned long long s1 = 56ull * 32ull;          // stride z  (elements)
    v4u_ g0;
    g0[0] = 1u;                                           // count=1, user mode
    g0[1] = lds_off;                                      // lds_addr
    g0[2] = (unsigned int)ga;                             // global_addr[31:0]
    g0[3] = ((unsigned int)(ga >> 32) & 0x01FFFFFFu) | 0x80000000u;  // addr[56:32] | type=2
    v8i_ g1;
    g1[0] = (int)(1u << 16);                              // data_size=2B
    g1[1] = (int)((tdim0 & 0xFFFFu) << 16);               // tensor_dim0[15:0] @ [63:48]
    g1[2] = (int)(((tdim0 >> 16) & 0xFFFFu) | ((tdim1 & 0xFFFFu) << 16));
    g1[3] = (int)(((tdim1 >> 16) & 0xFFFFu) | (32u << 16));   // tile_dim0=32
    g1[4] = (int)(14u | (14u << 16));                     // tile_dim1=14, tile_dim2=14
    g1[5] = (int)(unsigned int)(s0 & 0xFFFFFFFFull);      // dim0_stride[31:0]
    g1[6] = (int)(((unsigned int)(s0 >> 32) & 0xFFFFu) | (((unsigned int)s1 & 0xFFFFu) << 16));
    g1[7] = (int)(unsigned int)(s1 >> 16);                // dim1_stride[47:16]
    v4i_ g2;
    g2[0] = (int)tdim2;                                   // tensor_dim2
    g2[1] = 0; g2[2] = 0; g2[3] = 0;                      // dim3/tile_dim3 unused
    v4i_ g3 = {0, 0, 0, 0};
    v8i_ g4 = {0, 0, 0, 0, 0, 0, 0, 0};                   // extended group (unused)
    __builtin_amdgcn_tensor_load_to_lds(g0, g1, g2, g3, g4, 0);
  }
  // zero the padded rows 196..223 (outside the 14x14 tile)
  for (int s = 196 + tid; s < NN; s += 128) {
    v8bf z = {};
    *(v8bf*)(Ks + s * 32)      = z; *(v8bf*)(Ks + s * 32 + 8)  = z;
    *(v8bf*)(Ks + s * 32 + 16) = z; *(v8bf*)(Ks + s * 32 + 24) = z;
  }
#else
  for (int s = tid; s < NN; s += 128) {
    int i = s / 14, j = s - (s / 14) * 14;
    bool valid = (i < 14) && (row_lo + i <= 55) && (col_lo + j <= 55);
    int r = imin(row_lo + i, 55), c = imin(col_lo + j, 55);
    const bf16* ksrc = kbase + (size_t)(r * 56 + c) * 32;
#if HAVE_ASYNC_LDS
    if (valid) {
      // 64B row via 4 async DMA transfers (ASYNCcnt), no VGPR round-trip.
      const __attribute__((address_space(1))) void* g =
          (const __attribute__((address_space(1))) void*)ksrc;
      __attribute__((address_space(3))) void* l =
          (__attribute__((address_space(3))) void*)(Ks + s * 32);
      __builtin_amdgcn_global_load_async_to_lds_b128(g, l, 0, 0);
      __builtin_amdgcn_global_load_async_to_lds_b128(g, l, 16, 0);
      __builtin_amdgcn_global_load_async_to_lds_b128(g, l, 32, 0);
      __builtin_amdgcn_global_load_async_to_lds_b128(g, l, 48, 0);
    } else {
      v8bf z = {};
      *(v8bf*)(Ks + s * 32)      = z; *(v8bf*)(Ks + s * 32 + 8)  = z;
      *(v8bf*)(Ks + s * 32 + 16) = z; *(v8bf*)(Ks + s * 32 + 24) = z;
    }
#else
    v8bf z = {};
    v8bf k0 = z, k1 = z, k2 = z, k3 = z;
    if (valid) {
      k0 = *(const v8bf*)(ksrc);      k1 = *(const v8bf*)(ksrc + 8);
      k2 = *(const v8bf*)(ksrc + 16); k3 = *(const v8bf*)(ksrc + 24);
    }
    *(v8bf*)(Ks + s * 32)      = k0; *(v8bf*)(Ks + s * 32 + 8)  = k1;
    *(v8bf*)(Ks + s * 32 + 16) = k2; *(v8bf*)(Ks + s * 32 + 24) = k3;
#endif
  }
#endif

  // ---- V slab staging (transposed, d-major) ------------------------------
  for (int s = tid; s < NN; s += 128) {
    int i = s / 14, j = s - (s / 14) * 14;
    bool valid = (i < 14) && (row_lo + i <= 55) && (col_lo + j <= 55);
    int r = imin(row_lo + i, 55), c = imin(col_lo + j, 55);
    const bf16* vsrc = vbase + (size_t)(r * 56 + c) * 32;
    v8bf z = {};
    v8bf w0 = z, w1 = z, w2 = z, w3 = z;
    if (valid) {
      w0 = *(const v8bf*)(vsrc);      w1 = *(const v8bf*)(vsrc + 8);
      w2 = *(const v8bf*)(vsrc + 16); w3 = *(const v8bf*)(vsrc + 24);
    }
#pragma unroll
    for (int d = 0; d < 8; ++d) {
      Vst[d * NN + s]        = w0[d];
      Vst[(8 + d) * NN + s]  = w1[d];
      Vst[(16 + d) * NN + s] = w2[d];
      Vst[(24 + d) * NN + s] = w3[d];
    }
  }

#if HAVE_TDM
  if (wave == 0) __builtin_amdgcn_s_wait_tensorcnt(0);
#elif HAVE_ASYNC_LDS
#if __has_builtin(__builtin_amdgcn_s_wait_asynccnt)
  __builtin_amdgcn_s_wait_asynccnt(0);
#else
  asm volatile("s_wait_asynccnt 0x0" ::: "memory");
#endif
#endif
  __syncthreads();

  // GEMM1: S(64 x 224) = Q(64 x 32) @ Kslab^T   (wave w owns rows w*16..w*16+15)
  {
    int p = wave * 16 + nl;
    int px = x0 + (p >> 3), py = y0 + (p & 7);
    const bf16* qrow = qbase + (size_t)(px * 56 + py) * 32;
    v16bf aq = load_a_bf(qrow + half * 16);
#pragma unroll
    for (int nt = 0; nt < 14; ++nt) {
      int n = nt * 16 + nl;
      v16bf bk = load16bf(Ks + n * 32 + half * 16);
      v8f s = {};
      s = WMMA_BF16(aq, bk, s);
#pragma unroll
      for (int r = 0; r < 8; r++)
        Ss[(size_t)(wave * 16 + r + half * 8) * NN + n] = s[r];
    }
  }
  __syncthreads();

  // Masked softmax (49 valid neighbors per row) with relative position bias.
  if (tid < 64) {
    const int p = tid;
    const int px = x0 + (p >> 3), py = y0 + (p & 7);
    const int si = imin(imax(px - 3, 0), 49);
    const int sj = imin(imax(py - 3, 0), 49);
    const int si0 = si - row_lo, sj0 = sj - col_lo;
    const int ri0 = si - px + 6, rj0 = sj - py + 6;
    float* srow = Ss + (size_t)p * NN;
    const float* rp = rpb + h * 169;
    float mx = -3.0e38f;
#pragma unroll
    for (int i = 0; i < 7; i++)
#pragma unroll
      for (int j = 0; j < 7; j++) {
        int n = (si0 + i) * 14 + sj0 + j;
        float t = srow[n] + rp[(ri0 + i) * 13 + rj0 + j];
        srow[n] = t;
        mx = fmaxf(mx, t);
      }
    float sum = 0.f;
#pragma unroll
    for (int i = 0; i < 7; i++)
#pragma unroll
      for (int j = 0; j < 7; j++) {
        int n = (si0 + i) * 14 + sj0 + j;
        float e = __expf(srow[n] - mx);
        srow[n] = e;
        sum += e;
      }
    float inv = 1.0f / sum;
    bf16* prow = Ps + (size_t)p * NN;
    v8bf z = {};
#pragma unroll
    for (int n = 0; n < NN; n += 8) *(v8bf*)(prow + n) = z;
#pragma unroll
    for (int i = 0; i < 7; i++)
#pragma unroll
      for (int j = 0; j < 7; j++) {
        int n = (si0 + i) * 14 + sj0 + j;
        prow[n] = (bf16)(srow[n] * inv);
      }
  }
  __syncthreads();

  // GEMM2: O(64 x 32) = P(64 x 224) @ Vslab(224 x 32)
  v8f o0 = {}, o1 = {};
  const bf16* prow = Ps + (size_t)(wave * 16 + nl) * NN;
#pragma unroll
  for (int kt = 0; kt < 7; ++kt) {
    v16bf pa = load_a_bf(prow + kt * 32 + half * 16);
    v16bf b0 = load16bf(Vst + (size_t)nl * NN + kt * 32 + half * 16);
    v16bf b1 = load16bf(Vst + (size_t)(16 + nl) * NN + kt * 32 + half * 16);
    o0 = WMMA_BF16(pa, b0, o0);
    o1 = WMMA_BF16(pa, b1, o1);
  }
#pragma unroll
  for (int r = 0; r < 8; r++) {
    int p = wave * 16 + r + half * 8;
    int px = x0 + (p >> 3), py = y0 + (p & 7);
    size_t o = ((size_t)b * 3136 + px * 56 + py) * 256 + h * 32;
    attn_out[o + nl]      = (bf16)o0[r];
    attn_out[o + 16 + nl] = (bf16)o1[r];
  }
}

// ---------------------------------------------------------------------------
// Kernel 3: out = attn_out(bf16) @ proj_w^T + proj_b, fp32 output.
// ---------------------------------------------------------------------------
__global__ __launch_bounds__(32)
void proj_gemm_kernel(const bf16* __restrict__ ain, const float* __restrict__ W,
                      const float* __restrict__ bias, float* __restrict__ out) {
  const int lane = threadIdx.x & 31;
  const int nl = lane & 15, half = lane >> 4;
  const int row0 = blockIdx.x * 16;
  const int col0 = blockIdx.y * 16;
  const bf16* arow = ain + (size_t)(row0 + nl) * 256;
  const float* brow = W + (size_t)(col0 + nl) * 256;

  v8f acc = {};
#pragma unroll
  for (int k0 = 0; k0 < 256; k0 += 32) {
    v16bf a = load_a_bf(arow + k0 + half * 16);
    v16bf b = load_b_f32(brow + k0 + half * 16);
    acc = WMMA_BF16(a, b, acc);
  }
  const float bval = bias[col0 + nl];
#pragma unroll
  for (int r = 0; r < 8; r++)
    out[(size_t)(row0 + r + half * 8) * 256 + col0 + nl] = acc[r] + bval;
}

// ---------------------------------------------------------------------------
extern "C" void kernel_launch(void* const* d_in, const int* in_sizes, int n_in,
                              void* d_out, int out_size, void* d_ws, size_t ws_size,
                              hipStream_t stream) {
  const float* xq     = (const float*)d_in[0];
  const float* xkv    = (const float*)d_in[1];
  const float* qkv_w  = (const float*)d_in[2];
  const float* qkv_b  = (const float*)d_in[3];
  const float* rpb    = (const float*)d_in[4];
  const float* proj_w = (const float*)d_in[5];
  const float* proj_b = (const float*)d_in[6];
  float* out = (float*)d_out;

  const size_t NELT = (size_t)2 * 8 * 3136 * 32;  // per q/k/v/attn buffer
  bf16* qb = (bf16*)d_ws;
  bf16* kb = qb + NELT;
  bf16* vb = kb + NELT;
  bf16* ao = vb + NELT;

  // 1) QKV projection: 392 row tiles x 16 col tiles x {q,k,v}
  qkv_gemm_kernel<<<dim3(392, 16, 3), 32, 0, stream>>>(xq, xkv, qkv_w, qkv_b, qb, kb, vb);

  // 2) Neighborhood attention: 2*8*49 blocks, 4 waves each, 112 KB dynamic LDS
  const size_t smem_bytes = (size_t)NN * 32 * 2   // Ks
                          + (size_t)32 * NN * 2   // Vst
                          + (size_t)64 * NN * 4   // Ss
                          + (size_t)64 * NN * 2;  // Ps
  natten_kernel<<<dim3(784), 128, smem_bytes, stream>>>(qb, kb, vb, rpb, ao);

  // 3) Output projection
  proj_gemm_kernel<<<dim3(392, 16), 32, 0, stream>>>(ao, proj_w, proj_b, out);
}
